// DAC_33328946217038
// MI455X (gfx1250) — matrix operand order repaired and checked by reading, not compiled, not run on hardware
//
#include <hip/hip_runtime.h>
#include <cstdint>
#include <cstddef>

// ============================================================================
// DAC codec forward for MI455X (gfx1250, wave32, WMMA).
//
// All convs with IC%32==0 && OC%32==0 run as implicit GEMM on
// v_wmma_f32_16x16x32_f16: f16 channels-last activations, f32 accumulation,
// snake fused into the A-fragment load (per-channel alpha and 1/(a+eps)
// staged LDS -> hoisted to registers once per ic-chunk, so the inner loop is
// loads + sin/fma + 2 WMMAs), bias/residual fused in the epilogue. Each wave
// computes a 16(t) x 32(oc) tile = 2 WMMAs per A fragment to amortize A loads
// + snake VALU. Weights are device-packed into the WMMA B-fragment layout
// [k][ic/32][oc][32] (~154MB f16 ~= resident in 192MB L2). Transposed convs
// are lhs-dilated convs through the same WMMA path (stride is a power of two
// -> phase test / index via mask+shift, no integer division). VQ (dim-8) +
// edge convs (IC=1 / OC=1) are cheap VALU kernels.
// ============================================================================

typedef __attribute__((ext_vector_type(16))) _Float16 v16h;
typedef __attribute__((ext_vector_type(8)))  _Float16 v8h;
typedef __attribute__((ext_vector_type(8)))  float    v8f;

union H8 { v8h h; float4 f; };
union H16 { v16h h; float4 f[2]; };

// snake with precomputed reciprocal: x + sin(a*x)^2 * rinv
static __device__ __forceinline__ float snake_r(float x, float a, float rinv) {
  float s = __sinf(a * x);
  return __builtin_fmaf(s * s, rinv, x);
}

// ---------------------------------------------------------------------------
// Weight packing: torch conv w [OC][IC][K] (f32) -> WMMA-B layout
// [k][IC/32][OC][32] (f16).  B element (K=ic, N=oc) == W[oc][ic][k].
// ---------------------------------------------------------------------------
__global__ void pack_conv_weights(const float* __restrict__ W,
                                  _Float16* __restrict__ Wp,
                                  int OC, int IC, int Ksz) {
  size_t i = (size_t)blockIdx.x * blockDim.x + threadIdx.x;
  size_t total = (size_t)Ksz * IC * OC;
  if (i >= total) return;
  int l  = (int)(i & 31);
  size_t r = i >> 5;
  int oc = (int)(r % OC); r /= OC;
  int icc = (int)(r % (IC / 32));
  int k   = (int)(r / (IC / 32));
  int ic = icc * 32 + ((l & 7) | (((l >> 4) & 1) << 3) | (((l >> 3) & 1) << 4));
  Wp[i] = (_Float16)W[((size_t)oc * IC + ic) * Ksz + k];
}

// torch ConvTranspose1d w [IC][OC][K] -> effective conv weight
// wf[oc][ic][k] = w[ic][oc][K-1-k], packed same as above.
__global__ void pack_tconv_weights(const float* __restrict__ W,
                                   _Float16* __restrict__ Wp,
                                   int OC, int IC, int Ksz) {
  size_t i = (size_t)blockIdx.x * blockDim.x + threadIdx.x;
  size_t total = (size_t)Ksz * IC * OC;
  if (i >= total) return;
  int l  = (int)(i & 31);
  size_t r = i >> 5;
  int oc = (int)(r % OC); r /= OC;
  int icc = (int)(r % (IC / 32));
  int k   = (int)(r / (IC / 32));
  int ic = icc * 32 + ((l & 7) | (((l >> 4) & 1) << 3) | (((l >> 3) & 1) << 4));
  Wp[i] = (_Float16)W[((size_t)ic * OC + oc) * Ksz + (Ksz - 1 - k)];
}

// ---------------------------------------------------------------------------
// Core implicit-GEMM conv1d on WMMA. X:[B,Tin,IC] f16, Y:[B,Tout,OC] f16.
// One wave -> one 16(t) x 32(oc) tile (2 WMMAs per A fragment); 4 waves/block
// cover 64 t-rows. Loop nest: ic-chunk OUTER (alpha regs hoisted), k INNER.
// ---------------------------------------------------------------------------
template <bool PRE_SNAKE, bool RESIDUAL>
__global__ void __launch_bounds__(128)
conv1d_wmma(const _Float16* __restrict__ X, const _Float16* __restrict__ Wp,
            const float* __restrict__ bias, const float* __restrict__ alpha,
            const _Float16* __restrict__ Res, _Float16* __restrict__ Y,
            int Bn, int Tin, int Tout, int IC, int OC,
            int Ksz, int stride, int pad, int dil) {
  extern __shared__ float lds_a[];  // [0,IC): alpha, [IC,2IC): 1/(alpha+eps)
  if (PRE_SNAKE) {
    for (int i = threadIdx.x; i < IC; i += blockDim.x) {
      float a = alpha[i];
      lds_a[i] = a;
      lds_a[IC + i] = 1.0f / (a + 1e-9f);
    }
    __syncthreads();
  }
  const int lane = threadIdx.x & 31;
  const int wave = threadIdx.x >> 5;
  const int tTiles = (Tout + 63) >> 6;
  const int b  = blockIdx.x / tTiles;
  const int t0 = (blockIdx.x - b * tTiles) * 64 + wave * 16;
  const int oc0 = blockIdx.y * 32;
  if (t0 >= Tout) return;  // uniform per wave

  const int row  = lane & 15;     // A: M row; B/C: N column
  const int hsel = lane >> 4;     // which K-half this lane holds
  const int t = t0 + row;
  const bool rowValid = (t < Tout);
  const int icChunks = IC >> 5;
  const size_t kStride = (size_t)icChunks * OC * 32;  // Wp elems per k-tap
  const int tiBase = t * stride - pad;                // hoisted
  v8f acc0 = {}, acc1 = {};

  for (int c = 0; c < icChunks; ++c) {
    const int base = c * 32 + hsel * 8;
    float af[16], rf[16];
    if (PRE_SNAKE) {
#pragma unroll
      for (int e = 0; e < 8; ++e) {
        af[e]     = lds_a[base + e];
        rf[e]     = lds_a[IC + base + e];
        af[8 + e] = lds_a[base + 16 + e];
        rf[8 + e] = lds_a[IC + base + 16 + e];
      }
    }
    const _Float16* wbase =
        Wp + ((size_t)c * OC + (oc0 + row)) * 32 + hsel * 16;
    for (int k = 0; k < Ksz; ++k) {
      int ti = tiBase + k * dil;
      const bool av = rowValid && ti >= 0 && ti < Tin;
      const _Float16* xp =
          X + ((size_t)b * Tin + (size_t)(av ? ti : 0)) * IC + base;
      v16h a = {};
      if (av) {
        H8 u0, u1;
        u0.f = *(const float4*)xp;          // K ids base..base+7
        u1.f = *(const float4*)(xp + 16);   // K ids base+16..+23
        if (PRE_SNAKE) {
#pragma unroll
          for (int e = 0; e < 8; ++e) {
            u0.h[e] = (_Float16)snake_r((float)u0.h[e], af[e], rf[e]);
            u1.h[e] = (_Float16)snake_r((float)u1.h[e], af[8 + e], rf[8 + e]);
          }
        }
#pragma unroll
        for (int e = 0; e < 8; ++e) { a[e] = u0.h[e]; a[8 + e] = u1.h[e]; }
      }
      // B fragments for oc tiles oc0 and oc0+16
      const _Float16* wrow = wbase + (size_t)k * kStride;
      H16 ub0, ub1;
      ub0.f[0] = ((const float4*)wrow)[0];
      ub0.f[1] = ((const float4*)wrow)[1];
      ub1.f[0] = ((const float4*)(wrow + 16 * 32))[0];
      ub1.f[1] = ((const float4*)(wrow + 16 * 32))[1];
      acc0 = __builtin_amdgcn_wmma_f32_16x16x32_f16(
          false, a, false, ub0.h, (short)0, acc0, false, false);
      acc1 = __builtin_amdgcn_wmma_f32_16x16x32_f16(
          false, a, false, ub1.h, (short)0, acc1, false, false);
    }
  }

  const int oc = oc0 + row;
  const float bs0 = bias ? bias[oc] : 0.0f;
  const float bs1 = bias ? bias[oc + 16] : 0.0f;
#pragma unroll
  for (int v = 0; v < 8; ++v) {
    int tt = t0 + v + hsel * 8;     // C layout: M = v (+8 for lanes 16..31)
    if (tt < Tout) {
      size_t o = ((size_t)b * Tout + tt) * OC + oc;
      float r0 = acc0[v] + bs0;
      float r1 = acc1[v] + bs1;
      if (RESIDUAL) {
        r0 += (float)Res[o];
        r1 += (float)Res[o + 16];
      }
      Y[o] = (_Float16)r0;
      Y[o + 16] = (_Float16)r1;
    }
  }
}

// ---------------------------------------------------------------------------
// Transposed conv as lhs-dilated conv on the same WMMA pipeline (snake fused).
// y[t] = sum_k wf[k] * x[(t-padl+k)>>sshift]  when (t-padl+k) & (s-1) == 0.
// stride is a power of two -> mask/shift, no integer division.
// ---------------------------------------------------------------------------
__global__ void __launch_bounds__(128)
tconv1d_wmma(const _Float16* __restrict__ X, const _Float16* __restrict__ Wp,
             const float* __restrict__ bias, const float* __restrict__ alpha,
             _Float16* __restrict__ Y,
             int Bn, int Tin, int Tout, int IC, int OC,
             int Ksz, int sshift, int padl) {
  extern __shared__ float lds_a[];
  for (int i = threadIdx.x; i < IC; i += blockDim.x) {
    float a = alpha[i];
    lds_a[i] = a;
    lds_a[IC + i] = 1.0f / (a + 1e-9f);
  }
  __syncthreads();
  const int lane = threadIdx.x & 31;
  const int wave = threadIdx.x >> 5;
  const int tTiles = (Tout + 63) >> 6;
  const int b  = blockIdx.x / tTiles;
  const int t0 = (blockIdx.x - b * tTiles) * 64 + wave * 16;
  const int oc0 = blockIdx.y * 32;
  if (t0 >= Tout) return;

  const int row  = lane & 15;
  const int hsel = lane >> 4;
  const int t = t0 + row;
  const bool rowValid = (t < Tout);
  const int icChunks = IC >> 5;
  const size_t kStride = (size_t)icChunks * OC * 32;
  const int smask = (1 << sshift) - 1;
  const int qBase = t - padl;                  // hoisted
  v8f acc0 = {}, acc1 = {};

  for (int c = 0; c < icChunks; ++c) {
    const int base = c * 32 + hsel * 8;
    float af[16], rf[16];
#pragma unroll
    for (int e = 0; e < 8; ++e) {
      af[e]     = lds_a[base + e];
      rf[e]     = lds_a[IC + base + e];
      af[8 + e] = lds_a[base + 16 + e];
      rf[8 + e] = lds_a[IC + base + 16 + e];
    }
    const _Float16* wbase =
        Wp + ((size_t)c * OC + (oc0 + row)) * 32 + hsel * 16;
    for (int k = 0; k < Ksz; ++k) {
      const int q = qBase + k;
      const int ti = q >> sshift;              // valid only when q >= 0
      const bool av =
          rowValid && q >= 0 && (q & smask) == 0 && ti < Tin;
      const _Float16* xp =
          X + ((size_t)b * Tin + (size_t)(av ? ti : 0)) * IC + base;
      v16h a = {};
      if (av) {
        H8 u0, u1;
        u0.f = *(const float4*)xp;
        u1.f = *(const float4*)(xp + 16);
#pragma unroll
        for (int e = 0; e < 8; ++e) {
          u0.h[e] = (_Float16)snake_r((float)u0.h[e], af[e], rf[e]);
          u1.h[e] = (_Float16)snake_r((float)u1.h[e], af[8 + e], rf[8 + e]);
        }
#pragma unroll
        for (int e = 0; e < 8; ++e) { a[e] = u0.h[e]; a[8 + e] = u1.h[e]; }
      }
      const _Float16* wrow = wbase + (size_t)k * kStride;
      H16 ub0, ub1;
      ub0.f[0] = ((const float4*)wrow)[0];
      ub0.f[1] = ((const float4*)wrow)[1];
      ub1.f[0] = ((const float4*)(wrow + 16 * 32))[0];
      ub1.f[1] = ((const float4*)(wrow + 16 * 32))[1];
      acc0 = __builtin_amdgcn_wmma_f32_16x16x32_f16(
          false, a, false, ub0.h, (short)0, acc0, false, false);
      acc1 = __builtin_amdgcn_wmma_f32_16x16x32_f16(
          false, a, false, ub1.h, (short)0, acc1, false, false);
    }
  }
  const int oc = oc0 + row;
  const float bs0 = bias[oc];
  const float bs1 = bias[oc + 16];
#pragma unroll
  for (int v = 0; v < 8; ++v) {
    int tt = t0 + v + hsel * 8;
    if (tt < Tout) {
      size_t o = ((size_t)b * Tout + tt) * OC + oc;
      Y[o] = (_Float16)(acc0[v] + bs0);
      Y[o + 16] = (_Float16)(acc1[v] + bs1);
    }
  }
}

// ---------------------------------------------------------------------------
// Edge kernels (shapes WMMA can't fill).
// ---------------------------------------------------------------------------
__global__ void first_conv_kernel(const float* __restrict__ x,   // [B,1,T]
                                  const float* __restrict__ w,   // [64][1][7]
                                  const float* __restrict__ bias,
                                  _Float16* __restrict__ Y,      // [B,T,64]
                                  int Bn, int T) {
  int i = blockIdx.x * blockDim.x + threadIdx.x;
  if (i >= Bn * T * 64) return;
  int oc = i & 63;
  int t = (i >> 6) % T;
  int b = i / (T * 64);
  float acc = bias[oc];
  const float* xb = x + (size_t)b * T;
#pragma unroll
  for (int k = 0; k < 7; ++k) {
    int ti = t - 3 + k;
    if (ti >= 0 && ti < T) acc += xb[ti] * w[oc * 7 + k];
  }
  Y[((size_t)b * T + t) * 64 + oc] = (_Float16)acc;
}

__global__ void final_conv_tanh_kernel(const _Float16* __restrict__ X,  // [B,T,96]
                                       const float* __restrict__ alpha, // [96]
                                       const float* __restrict__ w,     // [1][96][7]
                                       const float* __restrict__ bias,  // [1]
                                       float* __restrict__ out,         // [B,1,T]
                                       int Bn, int T) {
  __shared__ float sw[96 * 7];
  __shared__ float sa[96];
  __shared__ float sr[96];
  for (int i = threadIdx.x; i < 96 * 7; i += blockDim.x) sw[i] = w[i];
  for (int i = threadIdx.x; i < 96; i += blockDim.x) {
    float a = alpha[i];
    sa[i] = a;
    sr[i] = 1.0f / (a + 1e-9f);
  }
  __syncthreads();
  int i = blockIdx.x * blockDim.x + threadIdx.x;
  if (i >= Bn * T) return;
  int b = i / T, t = i % T;
  float acc = bias[0];
  for (int k = 0; k < 7; ++k) {
    int ti = t - 3 + k;
    if (ti < 0 || ti >= T) continue;
    const _Float16* xr = X + ((size_t)b * T + ti) * 96;
    for (int ic = 0; ic < 96; ++ic)
      acc += snake_r((float)xr[ic], sa[ic], sr[ic]) * sw[ic * 7 + k];
  }
  out[(size_t)b * T + t] = tanhf(acc);
}

__global__ void f16_to_f32_kernel(const _Float16* __restrict__ X,
                                  float* __restrict__ Y, int n) {
  int i = blockIdx.x * blockDim.x + threadIdx.x;
  if (i < n) Y[i] = (float)X[i];
}

__global__ void zero_f32_kernel(float* __restrict__ p, int n) {
  int i = blockIdx.x * blockDim.x + threadIdx.x;
  if (i < n) p[i] = 0.0f;
}

// ---------------------------------------------------------------------------
// Residual VQ stage kernels (f32 VALU; < 1 GFLOP total).
// ---------------------------------------------------------------------------
__global__ void vq_project_in_kernel(const float* __restrict__ resid, // [B,T,1024]
                                     const float* __restrict__ wi,    // [8][1024]
                                     const float* __restrict__ bi,    // [8]
                                     float* __restrict__ ze,          // [B,T,8]
                                     float* __restrict__ lat,         // [B,72,T]
                                     int Bn, int T, int qi) {
  int i = blockIdx.x * blockDim.x + threadIdx.x;
  if (i >= Bn * T * 8) return;
  int c = i & 7;
  int r = i >> 3;
  int b = r / T, t = r % T;
  const float* x = resid + (size_t)r * 1024;
  const float* wr = wi + (size_t)c * 1024;
  float acc = bi[c];
  for (int ic = 0; ic < 1024; ++ic) acc += x[ic] * wr[ic];
  ze[(size_t)r * 8 + c] = acc;
  lat[((size_t)b * 72 + qi * 8 + c) * T + t] = acc;
}

__global__ void vq_nearest_kernel(const float* __restrict__ ze,  // [B,T,8]
                                  const float* __restrict__ cb,  // [1024][8]
                                  int* __restrict__ idx,
                                  float* __restrict__ codes,     // [B,9,T]
                                  int Bn, int T, int qi) {
  int r = blockIdx.x * blockDim.x + threadIdx.x;
  if (r >= Bn * T) return;
  float e[8];
  float n2 = 0.0f;
#pragma unroll
  for (int c = 0; c < 8; ++c) { e[c] = ze[(size_t)r * 8 + c]; n2 += e[c] * e[c]; }
  float inv = 1.0f / fmaxf(sqrtf(n2), 1e-12f);
  float en2 = 0.0f;
#pragma unroll
  for (int c = 0; c < 8; ++c) { e[c] *= inv; en2 += e[c] * e[c]; }
  float best = 3.4e38f;
  int bi = 0;
  for (int j = 0; j < 1024; ++j) {
    const float* cr = cb + (size_t)j * 8;
    float cn2 = 0.0f, dot = 0.0f;
#pragma unroll
    for (int c = 0; c < 8; ++c) cn2 += cr[c] * cr[c];
    float cinv = 1.0f / fmaxf(sqrtf(cn2), 1e-12f);
    float cnn = 0.0f;
#pragma unroll
    for (int c = 0; c < 8; ++c) {
      float cn = cr[c] * cinv;
      dot += e[c] * cn;
      cnn += cn * cn;
    }
    float d = en2 - 2.0f * dot + cnn;
    if (d < best) { best = d; bi = j; }
  }
  idx[r] = bi;
  int b = r / T, t = r % T;
  codes[((size_t)b * 9 + qi) * T + t] = (float)bi;
}

__global__ void vq_out_update_kernel(const int* __restrict__ idx,
                                     const float* __restrict__ cb,  // [1024][8]
                                     const float* __restrict__ wo,  // [1024][8]
                                     const float* __restrict__ bo,  // [1024]
                                     float* __restrict__ zq_acc,    // [B,T,1024]
                                     float* __restrict__ resid,     // [B,T,1024]
                                     int Bn, int T) {
  int i = blockIdx.x * blockDim.x + threadIdx.x;
  if (i >= Bn * T * 1024) return;
  int oc = i & 1023;
  int r = i >> 10;
  const float* e = cb + (size_t)idx[r] * 8;
  const float* wr = wo + (size_t)oc * 8;
  float v = bo[oc];
#pragma unroll
  for (int c = 0; c < 8; ++c) v += e[c] * wr[c];
  zq_acc[i] += v;
  resid[i] -= v;
}

__global__ void vq_losses_kernel(const float* __restrict__ ze,
                                 const float* __restrict__ cb,
                                 const int* __restrict__ idx,
                                 float* __restrict__ commit_out,
                                 float* __restrict__ cbl_out,
                                 int Bn, int T) {
  int r = blockIdx.x * blockDim.x + threadIdx.x;
  if (r >= Bn * T) return;
  const float* e = cb + (size_t)idx[r] * 8;
  float s = 0.0f;
#pragma unroll
  for (int c = 0; c < 8; ++c) {
    float d = ze[(size_t)r * 8 + c] - e[c];
    s += d * d;
  }
  float scale = 1.0f / ((float)Bn * 8.0f * (float)T);
  atomicAdd(commit_out, s * scale);
  atomicAdd(cbl_out, s * scale);
}

__global__ void zq_finalize_kernel(const float* __restrict__ zq_acc, // [B,T,1024]
                                   float* __restrict__ zq_out,       // [B,1024,T]
                                   _Float16* __restrict__ zq_f16,    // [B,T,1024]
                                   int Bn, int T) {
  int i = blockIdx.x * blockDim.x + threadIdx.x;
  if (i >= Bn * T * 1024) return;
  int oc = i & 1023;
  int r = i >> 10;
  int b = r / T, t = r % T;
  float v = zq_acc[i];
  zq_out[((size_t)b * 1024 + oc) * T + t] = v;
  zq_f16[i] = (_Float16)v;
}

// ===========================================================================
// Host orchestration
// ===========================================================================
static inline int cdiv(int a, int b) { return (a + b - 1) / b; }
static inline int ilog2(int v) { int s = 0; while ((1 << s) < v) ++s; return s; }

struct Arena {
  char* p;
  size_t off;
  void* take(size_t bytes) {
    size_t o = (off + 255) & ~(size_t)255;
    off = o + bytes;
    return (void*)(p + o);
  }
};

static _Float16* pack_conv(Arena& A, const float* W, int OC, int IC, int K,
                           hipStream_t s) {
  size_t n = (size_t)K * IC * OC;
  _Float16* wp = (_Float16*)A.take(n * sizeof(_Float16));
  pack_conv_weights<<<dim3((unsigned)((n + 255) / 256)), dim3(256), 0, s>>>(
      W, wp, OC, IC, K);
  return wp;
}

static _Float16* pack_tconv(Arena& A, const float* W, int OC, int IC, int K,
                            hipStream_t s) {
  size_t n = (size_t)K * IC * OC;
  _Float16* wp = (_Float16*)A.take(n * sizeof(_Float16));
  pack_tconv_weights<<<dim3((unsigned)((n + 255) / 256)), dim3(256), 0, s>>>(
      W, wp, OC, IC, K);
  return wp;
}

static void launch_conv(const _Float16* X, const _Float16* Wp, const float* bias,
                        const float* alpha, const _Float16* Res, _Float16* Y,
                        int B, int Tin, int Tout, int IC, int OC, int K,
                        int stride, int pad, int dil, hipStream_t s) {
  dim3 grid((unsigned)(B * cdiv(Tout, 64)), (unsigned)(OC / 32));
  size_t lds = alpha ? (size_t)2 * IC * sizeof(float) : 0;
  if (alpha && Res)
    conv1d_wmma<true, true><<<grid, 128, lds, s>>>(X, Wp, bias, alpha, Res, Y, B,
                                                   Tin, Tout, IC, OC, K, stride,
                                                   pad, dil);
  else if (alpha)
    conv1d_wmma<true, false><<<grid, 128, lds, s>>>(X, Wp, bias, alpha, Res, Y,
                                                    B, Tin, Tout, IC, OC, K,
                                                    stride, pad, dil);
  else if (Res)
    conv1d_wmma<false, true><<<grid, 128, lds, s>>>(X, Wp, bias, alpha, Res, Y,
                                                    B, Tin, Tout, IC, OC, K,
                                                    stride, pad, dil);
  else
    conv1d_wmma<false, false><<<grid, 128, lds, s>>>(X, Wp, bias, alpha, Res, Y,
                                                     B, Tin, Tout, IC, OC, K,
                                                     stride, pad, dil);
}

static void launch_tconv(const _Float16* X, const _Float16* Wp,
                         const float* bias, const float* alpha, _Float16* Y,
                         int B, int Tin, int Tout, int IC, int OC, int K,
                         int stride, int padl, hipStream_t s) {
  dim3 grid((unsigned)(B * cdiv(Tout, 64)), (unsigned)(OC / 32));
  size_t lds = (size_t)2 * IC * sizeof(float);
  tconv1d_wmma<<<grid, 128, lds, s>>>(X, Wp, bias, alpha, Y, B, Tin, Tout, IC,
                                      OC, K, ilog2(stride), padl);
}

template <class T>
static void swp(T& a, T& b) { T t = a; a = b; b = t; }

struct RU { const float *a1, *a2, *b1, *b2, *w1, *w2; };

extern "C" void kernel_launch(void* const* d_in, const int* in_sizes, int n_in,
                              void* d_out, int out_size, void* d_ws,
                              size_t ws_size, hipStream_t stream) {
  (void)in_sizes; (void)n_in; (void)out_size; (void)ws_size;
  const int Bn = 4, T0 = 65536, Tz = 128;

  // ---- walk inputs in jax pytree order (dict keys sorted, lists in order) ----
  int cur = 0;
  auto F = [&]() -> const float* { return (const float*)d_in[cur++]; };

  const float* audio = F();

  // dec_params: alpha_f, b0, bf, blocks[4]{alpha, bt, rus[3]{a1,a2,b1,b2,w1,w2}, wt}, w0, wf
  const float* dec_alpha_f = F();
  const float* dec_b0 = F();
  const float* dec_bf = F();
  struct DecB { const float *alpha, *bt; RU ru[3]; const float* wt; } db[4];
  for (int i = 0; i < 4; ++i) {
    db[i].alpha = F();
    db[i].bt = F();
    for (int r = 0; r < 3; ++r)
      db[i].ru[r] = {F(), F(), F(), F(), F(), F()};
    db[i].wt = F();
  }
  const float* dec_w0 = F();
  const float* dec_wf = F();

  // enc_params: alpha_f, b0, bf, blocks[4]{alpha, b, rus[3]{...}, w}, w0, wf
  const float* enc_alpha_f = F();
  const float* enc_b0 = F();
  const float* enc_bf = F();
  struct EncB { const float *alpha, *b; RU ru[3]; const float* w; } eb[4];
  for (int i = 0; i < 4; ++i) {
    eb[i].alpha = F();
    eb[i].b = F();
    for (int r = 0; r < 3; ++r)
      eb[i].ru[r] = {F(), F(), F(), F(), F(), F()};
    eb[i].w = F();
  }
  const float* enc_w0 = F();
  const float* enc_wf = F();

  cur++;  // n_quantizers (device int scalar; value 9 by construction)

  struct VQP { const float *bi, *bo, *cb, *wi, *wo; } vq[9];
  for (int i = 0; i < 9; ++i) vq[i] = {F(), F(), F(), F(), F()};

  // ---- workspace carve ----
  Arena W{(char*)d_ws, 0};
  const size_t CAPE = (size_t)Bn * 65536 * 96;  // max activation elements
  _Float16* bufA = (_Float16*)W.take(CAPE * 2);
  _Float16* bufB = (_Float16*)W.take(CAPE * 2);
  _Float16* bufC = (_Float16*)W.take(CAPE * 2);
  _Float16* zbuf = (_Float16*)W.take((size_t)Bn * Tz * 1024 * 2);
  _Float16* zq16 = (_Float16*)W.take((size_t)Bn * Tz * 1024 * 2);
  float* resid = (float*)W.take((size_t)Bn * Tz * 1024 * 4);
  float* zq_acc = (float*)W.take((size_t)Bn * Tz * 1024 * 4);
  float* ze = (float*)W.take((size_t)Bn * Tz * 8 * 4);
  int* idx = (int*)W.take((size_t)Bn * Tz * 4);

  // ---- pack all WMMA conv weights (f16 B-fragment layout; ~154MB, ~L2-sized) ----
  const int encD[4] = {64, 128, 256, 512};
  const int encS[4] = {2, 4, 8, 8};
  const int dils[3] = {1, 3, 9};
  _Float16 *e_w1p[4][3], *e_w2p[4][3], *e_bwp[4];
  for (int i = 0; i < 4; ++i) {
    int d = encD[i];
    for (int r = 0; r < 3; ++r) {
      e_w1p[i][r] = pack_conv(W, eb[i].ru[r].w1, d, d, 7, stream);
      e_w2p[i][r] = pack_conv(W, eb[i].ru[r].w2, d, d, 1, stream);
    }
    e_bwp[i] = pack_conv(W, eb[i].w, 2 * d, d, 2 * encS[i], stream);
  }
  _Float16* e_wfp = pack_conv(W, enc_wf, 1024, 1024, 3, stream);

  const int decI[4] = {1536, 768, 384, 192};
  const int decO[4] = {768, 384, 192, 96};
  const int decS[4] = {8, 8, 4, 2};
  _Float16* d_w0p = pack_conv(W, dec_w0, 1536, 1024, 7, stream);
  _Float16 *d_wtp[4], *d_w1p[4][3], *d_w2p[4][3];
  for (int i = 0; i < 4; ++i) {
    d_wtp[i] = pack_tconv(W, db[i].wt, decO[i], decI[i], 2 * decS[i], stream);
    for (int r = 0; r < 3; ++r) {
      d_w1p[i][r] = pack_conv(W, db[i].ru[r].w1, decO[i], decO[i], 7, stream);
      d_w2p[i][r] = pack_conv(W, db[i].ru[r].w2, decO[i], decO[i], 1, stream);
    }
  }

  float* out_f = (float*)d_out;
  const size_t RECON_OFF = 0;                        // [4,1,65536]
  const size_t ZQ_OFF = (size_t)Bn * T0;             // [4,1024,128]
  const size_t CODES_OFF = ZQ_OFF + (size_t)Bn * 1024 * Tz;   // [4,9,128]
  const size_t LAT_OFF = CODES_OFF + (size_t)Bn * 9 * Tz;     // [4,72,128]
  const size_t COMMIT_OFF = LAT_OFF + (size_t)Bn * 72 * Tz;

  _Float16 *A = bufA, *Btmp = bufB, *C = bufC;

  // ======================= ENCODER =======================
  {
    int nt = Bn * T0 * 64;
    first_conv_kernel<<<dim3((unsigned)cdiv(nt, 256)), dim3(256), 0, stream>>>(
        audio, enc_w0, enc_b0, A, Bn, T0);
  }
  int T = T0;
  for (int i = 0; i < 4; ++i) {
    int d = encD[i];
    for (int r = 0; r < 3; ++r) {
      int dl = dils[r];
      launch_conv(A, e_w1p[i][r], eb[i].ru[r].b1, eb[i].ru[r].a1, nullptr, Btmp,
                  Bn, T, T, d, d, 7, 1, 3 * dl, dl, stream);
      launch_conv(Btmp, e_w2p[i][r], eb[i].ru[r].b2, eb[i].ru[r].a2, A, C, Bn,
                  T, T, d, d, 1, 1, 0, 1, stream);
      swp(A, C);
    }
    int s = encS[i];
    int Tout = T / s;
    launch_conv(A, e_bwp[i], eb[i].b, eb[i].alpha, nullptr, C, Bn, T, Tout, d,
                2 * d, 2 * s, s, (s + 1) / 2, 1, stream);
    swp(A, C);
    T = Tout;
  }
  // final: snake(alpha_f) + conv 1024->1024 k3 pad1 -> z
  launch_conv(A, e_wfp, enc_bf, enc_alpha_f, nullptr, zbuf, Bn, T, T, 1024,
              1024, 3, 1, 1, 1, stream);

  // ======================= RVQ (9 stages) =======================
  {
    int n = Bn * Tz * 1024;
    f16_to_f32_kernel<<<dim3((unsigned)cdiv(n, 256)), dim3(256), 0, stream>>>(
        zbuf, resid, n);
    zero_f32_kernel<<<dim3((unsigned)cdiv(n, 256)), dim3(256), 0, stream>>>(
        zq_acc, n);
    zero_f32_kernel<<<dim3(1), dim3(32), 0, stream>>>(out_f + COMMIT_OFF, 2);
    for (int q = 0; q < 9; ++q) {
      int np = Bn * Tz * 8;
      vq_project_in_kernel<<<dim3((unsigned)cdiv(np, 256)), dim3(256), 0,
                             stream>>>(resid, vq[q].wi, vq[q].bi, ze,
                                       out_f + LAT_OFF, Bn, Tz, q);
      int nr = Bn * Tz;
      vq_nearest_kernel<<<dim3((unsigned)cdiv(nr, 128)), dim3(128), 0, stream>>>(
          ze, vq[q].cb, idx, out_f + CODES_OFF, Bn, Tz, q);
      vq_out_update_kernel<<<dim3((unsigned)cdiv(n, 256)), dim3(256), 0,
                             stream>>>(idx, vq[q].cb, vq[q].wo, vq[q].bo,
                                       zq_acc, resid, Bn, Tz);
      vq_losses_kernel<<<dim3((unsigned)cdiv(nr, 128)), dim3(128), 0, stream>>>(
          ze, vq[q].cb, idx, out_f + COMMIT_OFF, out_f + COMMIT_OFF + 1, Bn,
          Tz);
    }
    zq_finalize_kernel<<<dim3((unsigned)cdiv(n, 256)), dim3(256), 0, stream>>>(
        zq_acc, out_f + ZQ_OFF, zq16, Bn, Tz);
  }

  // ======================= DECODER =======================
  launch_conv(zq16, d_w0p, dec_b0, nullptr, nullptr, A, Bn, Tz, Tz, 1024, 1536,
              7, 1, 3, 1, stream);
  T = Tz;
  for (int i = 0; i < 4; ++i) {
    int s = decS[i], din = decI[i], dout = decO[i];
    int K = 2 * s;
    int padl = 2 * s - 1 - (s + 1) / 2;
    int Tout = T * s;
    launch_tconv(A, d_wtp[i], db[i].bt, db[i].alpha, C, Bn, T, Tout, din, dout,
                 K, s, padl, stream);
    swp(A, C);
    T = Tout;
    for (int r = 0; r < 3; ++r) {
      int dl = dils[r];
      launch_conv(A, d_w1p[i][r], db[i].ru[r].b1, db[i].ru[r].a1, nullptr, Btmp,
                  Bn, T, T, dout, dout, 7, 1, 3 * dl, dl, stream);
      launch_conv(Btmp, d_w2p[i][r], db[i].ru[r].b2, db[i].ru[r].a2, A, C, Bn,
                  T, T, dout, dout, 1, 1, 0, 1, stream);
      swp(A, C);
    }
  }
  // final: snake(alpha_f) + conv 96->1 k7 pad3 + tanh -> recon
  {
    int nr = Bn * T;
    final_conv_tanh_kernel<<<dim3((unsigned)cdiv(nr, 128)), dim3(128), 0,
                             stream>>>(A, dec_alpha_f, dec_wf, dec_bf,
                                       out_f + RECON_OFF, Bn, T);
  }
}
